// GPSLayer_53360673686110
// MI455X (gfx1250) — matrix operand order repaired and checked by reading, not compiled, not run on hardware
//
#include <hip/hip_runtime.h>
#include <hip/hip_bf16.h>

// ---------------------------------------------------------------------------
// GPS (GraphGPS) layer for gfx1250: GIN local MPNN + global MHA + FFN, 3x BN.
// N=32768 nodes, D=128, E=524288 edges, B=64 graphs x L=512, H=8 heads x DH=16.
// GEMMs via V_WMMA_F32_16X16X32_BF16; weight tiles staged to LDS via the
// Tensor Data Mover (tensor_load_to_lds, TENSORcnt).
// ---------------------------------------------------------------------------

typedef __attribute__((ext_vector_type(16))) __bf16 v16bf;
typedef __attribute__((ext_vector_type(8)))  float  v8f;
typedef __attribute__((ext_vector_type(4)))  unsigned int v4u;
typedef __attribute__((ext_vector_type(8)))  int    v8i;
typedef __attribute__((ext_vector_type(4)))  int    v4i;

#define NODES 32768
#define DIM   128
#define EDGES 524288
#define GB    64
#define GL    512
#define HEADS 8
#define DH    16
#define TCOLS 64          // GEMM block column-tile width
#define KMAX  256         // largest K among the layer GEMMs

// ---------------------------------------------------------------------------
// Elementwise helpers
// ---------------------------------------------------------------------------
__global__ void zero_f32(float* __restrict__ p, int n4) {
    int i = blockIdx.x * blockDim.x + threadIdx.x;
    if (i < n4) ((float4*)p)[i] = make_float4(0.f, 0.f, 0.f, 0.f);
}

// agg[dst] += x[src]  (GIN neighbourhood sum), one thread = (edge, 4-float chunk)
__global__ void seg_scatter_add(const float* __restrict__ x,
                                const int* __restrict__ ei,
                                float* __restrict__ agg) {
    int idx = blockIdx.x * blockDim.x + threadIdx.x;      // E * (D/4) threads
    int e = idx >> 5;                                     // D/4 = 32 chunks
    int c = idx & 31;
    if (e >= EDGES) return;
    int src = ei[e];
    int dst = ei[EDGES + e];
    float4 val = ((const float4*)(x + (size_t)src * DIM))[c];
    float* out = agg + (size_t)dst * DIM + c * 4;
    atomicAdd(out + 0, val.x);
    atomicAdd(out + 1, val.y);
    atomicAdd(out + 2, val.z);
    atomicAdd(out + 3, val.w);
}

// h0 = (1+eps)*x + agg
__global__ void gin_fuse(const float* __restrict__ x, const float* __restrict__ agg,
                         const float* __restrict__ eps, float* __restrict__ h0, int n) {
    int i = blockIdx.x * blockDim.x + threadIdx.x;
    if (i < n) h0[i] = (1.0f + eps[0]) * x[i] + agg[i];
}

__global__ void add_f32(const float* __restrict__ a, const float* __restrict__ b,
                        float* __restrict__ c, int n) {
    int i = blockIdx.x * blockDim.x + threadIdx.x;
    if (i < n) c[i] = a[i] + b[i];
}

// ---------------------------------------------------------------------------
// BatchNorm (training-mode batch stats over node dimension)
// ---------------------------------------------------------------------------
__global__ void col_stats(const float* __restrict__ in, float* __restrict__ mean,
                          float* __restrict__ var) {
    const int d = blockIdx.x;                 // one block per column (DIM blocks)
    const int t = threadIdx.x;                // 256 threads
    float s = 0.f, ss = 0.f;
    for (int r = t; r < NODES; r += 256) {
        float xv = in[(size_t)r * DIM + d];
        s += xv; ss += xv * xv;
    }
    __shared__ float sh[256], sh2[256];
    sh[t] = s; sh2[t] = ss;
    __syncthreads();
    for (int o = 128; o > 0; o >>= 1) {
        if (t < o) { sh[t] += sh[t + o]; sh2[t] += sh2[t + o]; }
        __syncthreads();
    }
    if (t == 0) {
        float m = sh[0] * (1.0f / NODES);
        mean[d] = m;
        var[d] = sh2[0] * (1.0f / NODES) - m * m;
    }
}

__global__ void bn_apply(const float* __restrict__ in, const float* __restrict__ mean,
                         const float* __restrict__ var, const float* __restrict__ g,
                         const float* __restrict__ beta, float* __restrict__ out, int n) {
    int i = blockIdx.x * blockDim.x + threadIdx.x;
    if (i >= n) return;
    int d = i & (DIM - 1);
    out[i] = (in[i] - mean[d]) * __frsqrt_rn(var[d] + 1e-5f) * g[d] + beta[d];
}

// ---------------------------------------------------------------------------
// Generic WMMA GEMM: C[M,Nn] = act(A[M,K] @ W[K,Nn] + bias (+ res))
// Block = 8 waves; block tile = 128 rows x 64 cols; wave tile = 16 x 64.
// W tile (K x 64 fp32) staged into LDS once per block by the Tensor Data
// Mover (TDM); A fragments loaded as float4 pairs; bf16 WMMA, fp32 accum.
// ---------------------------------------------------------------------------
__global__ __launch_bounds__(256)
void wmma_gemm(const float* __restrict__ A, const float* __restrict__ W,
               const float* __restrict__ bias, const float* __restrict__ res,
               float* __restrict__ C, int Nn, int K, int act) {
    __shared__ float wtile[KMAX * TCOLS];                 // 64 KB
    const int tid  = threadIdx.x;
    const int lane = tid & 31;
    const int wave = tid >> 5;
    const int mr   = lane & 15;
    const int half = lane >> 4;

    // ---- stage W tile (rows 0..K-1, cols blockIdx.y*64..+63) via TDM ----
    if (wave == 0) {
        unsigned long long gaddr =
            (unsigned long long)(size_t)(W + (size_t)blockIdx.y * TCOLS);
        unsigned lds_base = (unsigned)(size_t)(&wtile[0]);
        v4u g0;
        g0[0] = 1u;                                       // count=1, user mode
        g0[1] = lds_base;                                 // lds_addr
        g0[2] = (unsigned)(gaddr & 0xFFFFFFFFull);        // global_addr lo
        g0[3] = (unsigned)((gaddr >> 32) & 0x01FFFFFFull) // global_addr hi
              | 0x80000000u;                              // type=2 ("image")
        v8i g1;
        g1[0] = 0x00020000;                               // data_size=4B, no mask
        g1[1] = (int)(((unsigned)Nn & 0xFFFFu) << 16);    // tensor_dim0[15:0]
        g1[2] = (int)((((unsigned)Nn >> 16) & 0xFFFFu)    // tensor_dim0[31:16]
              | ((unsigned)K << 16));                     // tensor_dim1[15:0]
        g1[3] = (int)((((unsigned)K >> 16) & 0xFFFFu)     // tensor_dim1[31:16]
              | ((unsigned)TCOLS << 16));                 // tile_dim0 = 64
        g1[4] = (int)((unsigned)K & 0xFFFFu);             // tile_dim1 = K, tile_dim2=0
        g1[5] = Nn;                                       // tensor_dim0_stride lo32
        g1[6] = 0;                                        // stride hi, dim1_stride lo
        g1[7] = 0;
        v4i gz = {0, 0, 0, 0};
#if defined(__clang_major__) && (__clang_major__ >= 23)
        v8i gz8 = {0, 0, 0, 0, 0, 0, 0, 0};
        __builtin_amdgcn_tensor_load_to_lds(g0, g1, gz, gz, gz8, 0);
#else
        __builtin_amdgcn_tensor_load_to_lds(g0, g1, gz, gz, 0);
#endif
        __builtin_amdgcn_s_wait_tensorcnt(0);
    }
    __syncthreads();

    const int rowBase = blockIdx.x * 128 + wave * 16;
    const int rowA    = rowBase + mr;                     // A-fragment row
    const float* pa   = A + (size_t)rowA * K;

    v8f c0 = {}, c1 = {}, c2 = {}, c3 = {};
    for (int kk = 0; kk < K; kk += 32) {
        if (kk + 32 < K) __builtin_prefetch(pa + kk + 32, 0, 1);   // global_prefetch_b8
        // A 16x32 bf16 fragment: two contiguous 8-float runs per lane
        const float4* a4lo = (const float4*)(pa + kk + half * 8);
        const float4* a4hi = (const float4*)(pa + kk + 16 + half * 8);
        float4 r0 = a4lo[0], r1 = a4lo[1];
        float4 r2 = a4hi[0], r3 = a4hi[1];
        v16bf a;
        a[0] = (__bf16)r0.x;  a[1] = (__bf16)r0.y;  a[2]  = (__bf16)r0.z;  a[3]  = (__bf16)r0.w;
        a[4] = (__bf16)r1.x;  a[5] = (__bf16)r1.y;  a[6]  = (__bf16)r1.z;  a[7]  = (__bf16)r1.w;
        a[8] = (__bf16)r2.x;  a[9] = (__bf16)r2.y;  a[10] = (__bf16)r2.z;  a[11] = (__bf16)r2.w;
        a[12] = (__bf16)r3.x; a[13] = (__bf16)r3.y; a[14] = (__bf16)r3.z;  a[15] = (__bf16)r3.w;
        // B 32x16 bf16 fragments from the LDS-staged W tile
        v16bf b0, b1, b2, b3;
#pragma unroll
        for (int e = 0; e < 16; ++e) {
            const float* wr = &wtile[(kk + (e < 8 ? half * 8 + e : 8 + half * 8 + e)) * TCOLS + mr];
            b0[e] = (__bf16)wr[0];
            b1[e] = (__bf16)wr[16];
            b2[e] = (__bf16)wr[32];
            b3[e] = (__bf16)wr[48];
        }
        c0 = __builtin_amdgcn_wmma_f32_16x16x32_bf16(false, a, false, b0, (short)0, c0, false, false);
        c1 = __builtin_amdgcn_wmma_f32_16x16x32_bf16(false, a, false, b1, (short)0, c1, false, false);
        c2 = __builtin_amdgcn_wmma_f32_16x16x32_bf16(false, a, false, b2, (short)0, c2, false, false);
        c3 = __builtin_amdgcn_wmma_f32_16x16x32_bf16(false, a, false, b3, (short)0, c3, false, false);
    }

    // C/D layout: VGPR r -> M = half*8 + r, N = lane&15
    const int colB = blockIdx.y * TCOLS + mr;
#pragma unroll
    for (int r = 0; r < 8; ++r) {
        int row = rowBase + half * 8 + r;
        float vv[4] = { c0[r], c1[r], c2[r], c3[r] };
#pragma unroll
        for (int s = 0; s < 4; ++s) {
            int col = colB + s * 16;
            float v = vv[s] + bias[col];
            if (res) v += res[(size_t)row * Nn + col];
            if (act) v = fmaxf(v, 0.f);
            C[(size_t)row * Nn + col] = v;
        }
    }
}

// ---------------------------------------------------------------------------
// Attention: per (query-tile, head, graph). One wave.
// scores tile (16 x 512) in LDS, softmax, P@V with bf16 WMMA (DH padded to K=32)
// ---------------------------------------------------------------------------
__global__ __launch_bounds__(32)
void attn_kernel(const float* __restrict__ q, const float* __restrict__ k,
                 const float* __restrict__ v, float* __restrict__ o) {
    const int lane = threadIdx.x;
    const int mr   = lane & 15;
    const int half = lane >> 4;
    const int qt = blockIdx.x;      // query tile 0..31
    const int h  = blockIdx.y;      // head
    const int b  = blockIdx.z;      // graph

    __shared__ float  sc[16 * GL];
    __shared__ __bf16 pb[16 * GL];

    // Q fragment, 16x16 padded to 16x32 (lanes 16..31 == K>=16 stay zero)
    v16bf aq = {};
    const int qnode = b * GL + qt * 16 + mr;
    const float* pq = q + (size_t)qnode * DIM + h * DH;
#pragma unroll
    for (int e = 0; e < 16; ++e) aq[e] = half ? (__bf16)0.f : (__bf16)pq[e];

    // scores = Q K^T / sqrt(DH)
    for (int kt = 0; kt < GL / 16; ++kt) {
        v16bf bkf = {};                      // lanes 16..31 carry K>=16 -> zero
        if (half == 0) {
            const float* pk = k + (size_t)(b * GL + kt * 16 + mr) * DIM + h * DH;
#pragma unroll
            for (int e = 0; e < 16; ++e) bkf[e] = (__bf16)pk[e];
        }
        v8f s = {};
        s = __builtin_amdgcn_wmma_f32_16x16x32_bf16(false, aq, false, bkf, (short)0, s, false, false);
#pragma unroll
        for (int r = 0; r < 8; ++r)
            sc[(half * 8 + r) * GL + kt * 16 + mr] = s[r] * 0.25f;   // 1/sqrt(16)
    }
    __syncthreads();

    // row softmax: one lane per query row
    if (lane < 16) {
        float mx = -1e30f;
        for (int j = 0; j < GL; ++j) mx = fmaxf(mx, sc[lane * GL + j]);
        float sum = 0.f;
        for (int j = 0; j < GL; ++j) {
            float e = __expf(sc[lane * GL + j] - mx);
            sc[lane * GL + j] = e;
            sum += e;
        }
        float inv = 1.0f / sum;
        for (int j = 0; j < GL; ++j) pb[lane * GL + j] = (__bf16)(sc[lane * GL + j] * inv);
    }
    __syncthreads();

    // O = P[16,512] @ V[512,16]
    v8f acc = {};
    for (int kk = 0; kk < GL; kk += 32) {
        v16bf ap;
#pragma unroll
        for (int e = 0; e < 8; ++e) ap[e]     = pb[mr * GL + kk + half * 8 + e];
#pragma unroll
        for (int e = 0; e < 8; ++e) ap[8 + e] = pb[mr * GL + kk + 16 + half * 8 + e];
        v16bf bvf;
#pragma unroll
        for (int e = 0; e < 16; ++e) {
            int kr = kk + (e < 8 ? half * 8 + e : 8 + half * 8 + e);
            bvf[e] = (__bf16)v[(size_t)(b * GL + kr) * DIM + h * DH + mr];
        }
        acc = __builtin_amdgcn_wmma_f32_16x16x32_bf16(false, ap, false, bvf, (short)0, acc, false, false);
    }
#pragma unroll
    for (int r = 0; r < 8; ++r) {
        int qn = b * GL + qt * 16 + half * 8 + r;
        o[(size_t)qn * DIM + h * DH + mr] = acc[r];
    }
}

// ---------------------------------------------------------------------------
// Host-side orchestration
// ---------------------------------------------------------------------------
extern "C" void kernel_launch(void* const* d_in, const int* in_sizes, int n_in,
                              void* d_out, int out_size, void* d_ws, size_t ws_size,
                              hipStream_t stream) {
    (void)in_sizes; (void)n_in; (void)out_size; (void)ws_size;
    const float* x      = (const float*)d_in[0];
    const int*   ei     = (const int*)  d_in[1];
    const float* eps    = (const float*)d_in[2];
    const float* W_loc1 = (const float*)d_in[3];
    const float* b_loc1 = (const float*)d_in[4];
    const float* W_loc2 = (const float*)d_in[5];
    const float* b_loc2 = (const float*)d_in[6];
    const float* g1     = (const float*)d_in[7];
    const float* beta1  = (const float*)d_in[8];
    const float* Wq     = (const float*)d_in[9];
    const float* bq     = (const float*)d_in[10];
    const float* Wk     = (const float*)d_in[11];
    const float* bk     = (const float*)d_in[12];
    const float* Wv     = (const float*)d_in[13];
    const float* bv     = (const float*)d_in[14];
    const float* Wo     = (const float*)d_in[15];
    const float* bo     = (const float*)d_in[16];
    const float* g2     = (const float*)d_in[17];
    const float* beta2  = (const float*)d_in[18];
    const float* Wf1    = (const float*)d_in[19];
    const float* bf1    = (const float*)d_in[20];
    const float* Wf2    = (const float*)d_in[21];
    const float* bf2    = (const float*)d_in[22];
    const float* g3     = (const float*)d_in[23];
    const float* beta3  = (const float*)d_in[24];
    float* out = (float*)d_out;

    // scratch pool: 8 units of N*D floats + BN stats
    const size_t UNIT = (size_t)NODES * DIM;
    float* U = (float*)d_ws;
    float* u0 = U + 0 * UNIT;   // agg -> hl_pre -> hsum
    float* u1 = U + 1 * UNIT;   // h0  -> h_local
    float* u2 = U + 2 * UNIT;   // t1(2 units) -> ha_pre -> f1(2 units)
    float* u4 = U + 4 * UNIT;   // q   -> h_attn
    float* u5 = U + 5 * UNIT;   // k   -> f_pre
    float* u6 = U + 6 * UNIT;   // v
    float* u7 = U + 7 * UNIT;   // attn_o
    float* mean = U + 8 * UNIT;
    float* var  = mean + 256;

    const int NE = NODES * DIM;                       // 4194304
    const int EB = 256;
    dim3 g256((NE + EB - 1) / EB);
    dim3 gemmGrid1(NODES / 128, 256 / TCOLS);         // K->256 GEMMs
    dim3 gemmGrid0(NODES / 128, 128 / TCOLS);         // K->128 GEMMs

    // --- GIN local branch ---
    zero_f32<<<dim3(NE / 4 / EB), EB, 0, stream>>>(u0, NE / 4);
    seg_scatter_add<<<dim3(EDGES * 32 / EB), EB, 0, stream>>>(x, ei, u0);
    gin_fuse<<<g256, EB, 0, stream>>>(x, u0, eps, u1, NE);
    wmma_gemm<<<gemmGrid1, 256, 0, stream>>>(u1, W_loc1, b_loc1, nullptr, u2, 256, 128, 1);
    wmma_gemm<<<gemmGrid0, 256, 0, stream>>>(u2, W_loc2, b_loc2, x, u0, 128, 256, 0);
    col_stats<<<dim3(DIM), 256, 0, stream>>>(u0, mean, var);
    bn_apply<<<g256, EB, 0, stream>>>(u0, mean, var, g1, beta1, u1, NE);  // h_local

    // --- global attention branch ---
    wmma_gemm<<<gemmGrid0, 256, 0, stream>>>(x, Wq, bq, nullptr, u4, 128, 128, 0);
    wmma_gemm<<<gemmGrid0, 256, 0, stream>>>(x, Wk, bk, nullptr, u5, 128, 128, 0);
    wmma_gemm<<<gemmGrid0, 256, 0, stream>>>(x, Wv, bv, nullptr, u6, 128, 128, 0);
    attn_kernel<<<dim3(GL / 16, HEADS, GB), 32, 0, stream>>>(u4, u5, u6, u7);
    wmma_gemm<<<gemmGrid0, 256, 0, stream>>>(u7, Wo, bo, x, u2, 128, 128, 0);
    col_stats<<<dim3(DIM), 256, 0, stream>>>(u2, mean, var);
    bn_apply<<<g256, EB, 0, stream>>>(u2, mean, var, g2, beta2, u4, NE);  // h_attn

    // --- combine + FFN ---
    add_f32<<<g256, EB, 0, stream>>>(u1, u4, u0, NE);                     // hsum
    wmma_gemm<<<gemmGrid1, 256, 0, stream>>>(u0, Wf1, bf1, nullptr, u2, 256, 128, 1);
    wmma_gemm<<<gemmGrid0, 256, 0, stream>>>(u2, Wf2, bf2, u0, u5, 128, 256, 0);
    col_stats<<<dim3(DIM), 256, 0, stream>>>(u5, mean, var);
    bn_apply<<<g256, EB, 0, stream>>>(u5, mean, var, g3, beta3, out, NE);
}
